// LSTM_72206990180809
// MI455X (gfx1250) — matrix operand order repaired and checked by reading, not compiled, not run on hardware
//
#include <hip/hip_runtime.h>
#include <hip/hip_bf16.h>
#include <math.h>

// LSTM LM forward for MI455X (gfx1250, wave32, WMMA).
// B=64, T=512, V=32000, E=512, H=1024, 4H=4096.

typedef __attribute__((ext_vector_type(16))) __bf16 bf16x16;
typedef __attribute__((ext_vector_type(8)))  float  v8f;

static constexpr int kB  = 64;
static constexpr int kT  = 512;
static constexpr int kE  = 512;
static constexpr int kH  = 1024;
static constexpr int k4H = 4096;
static constexpr int kNWG_REC = 64;   // one WG per 16 hidden units

__device__ __forceinline__ v8f vzero8() {
  v8f z;
#pragma unroll
  for (int i = 0; i < 8; ++i) z[i] = 0.0f;
  return z;
}

// Build a WMMA A/B bf16 fragment (16 values for this lane) from f32 memory.
// Lane layout (ISA 16-bit A 16x32): base=(lane>>4)*8; elems i -> K=base+i,
// elems 8+i -> K=16+base+i.  Caller passes base+off already including +base.
__device__ __forceinline__ bf16x16 cvt_gather16_f32(const float* base, int off) {
  const float* p = base + off;
  bf16x16 v;
#pragma unroll
  for (int i = 0; i < 8; ++i) {
    v[i]     = (__bf16)p[i];
    v[i + 8] = (__bf16)p[i + 16];
  }
  return v;
}

// Same fragment from pre-converted bf16 memory (two 16B loads).
__device__ __forceinline__ bf16x16 load16_bf16(const __bf16* p) {
  union { bf16x16 v; uint4 q[2]; } u;
  u.q[0] = *(const uint4*)(p);
  u.q[1] = *(const uint4*)(p + 16);
  return u.v;
}

__device__ __forceinline__ float sigmoidf(float x) {
  return 1.0f / (1.0f + __expf(-x));
}

// -------------------------------------------------------------------------
// Kernel 1: xproj[t,b,:] = emb[x[b,t]] @ W_ih^T + b_ih + b_hh   (f32 out)
// Row m = t*64 + b  (M = 32768, N = 4096, K = 512), bf16 WMMA, f32 acc.
// Block = 256 thr (8 waves). Wave tile 32x64 -> block tile 64x256.
// A/B addressed via 32-bit element offsets (GVS form: SGPR base + v-offset)
// to keep VGPR pressure below the spill threshold.
// -------------------------------------------------------------------------
__global__ void lstm_xproj_gemm(const int* __restrict__ x,
                                const float* __restrict__ emb,
                                const float* __restrict__ W_ih,
                                const float* __restrict__ b_ih,
                                const float* __restrict__ b_hh,
                                float* __restrict__ xp) {
  const int lane  = threadIdx.x & 31;
  const int wave  = threadIdx.x >> 5;
  const int wm    = wave & 1;
  const int wn    = wave >> 1;
  const int rowBase = blockIdx.x * 64 + wm * 32;
  const int colBase = blockIdx.y * 256 + wn * 64;
  const int lr    = lane & 15;
  const int half8 = (lane >> 4) * 8;

  // Fixed per-lane 32-bit element offsets (embedding gather hoisted).
  int aoff[2];
#pragma unroll
  for (int mb = 0; mb < 2; ++mb) {
    int m = rowBase + mb * 16 + lr;           // m = t*64 + b
    int b = m & 63;
    int t = m >> 6;
    int er = x[b * kT + t];
    aoff[mb] = er * kE + half8;               // < 16.4M elements, fits int
  }
  int boff[4];
#pragma unroll
  for (int nb = 0; nb < 4; ++nb) {
    int n = colBase + nb * 16 + lr;
    boff[nb] = n * kE + half8;                // B[k,n] = W_ih[n*E + k]
  }

  v8f acc[2][4];
#pragma unroll
  for (int mb = 0; mb < 2; ++mb)
#pragma unroll
    for (int nb = 0; nb < 4; ++nb) acc[mb][nb] = vzero8();

  for (int kb = 0; kb < kE; kb += 32) {
    bf16x16 afr[2], bfr[4];
#pragma unroll
    for (int mb = 0; mb < 2; ++mb) afr[mb] = cvt_gather16_f32(emb, aoff[mb] + kb);
#pragma unroll
    for (int nb = 0; nb < 4; ++nb) bfr[nb] = cvt_gather16_f32(W_ih, boff[nb] + kb);
#pragma unroll
    for (int mb = 0; mb < 2; ++mb)
#pragma unroll
      for (int nb = 0; nb < 4; ++nb)
        acc[mb][nb] = __builtin_amdgcn_wmma_f32_16x16x32_bf16(
            false, afr[mb], false, bfr[nb], (short)0, acc[mb][nb],
            false, false);
  }

  // C/D layout: vec elem r <-> M = mb*16 + 8*(lane>>4) + r, N = lane&15.
  // xp is 512 MB streamed once -> non-temporal stores (keep L2 for weights).
#pragma unroll
  for (int nb = 0; nb < 4; ++nb) {
    int n = colBase + nb * 16 + lr;
    float bias = b_ih[n] + b_hh[n];
#pragma unroll
    for (int mb = 0; mb < 2; ++mb) {
      int m0 = rowBase + mb * 16 + half8;
#pragma unroll
      for (int r = 0; r < 8; ++r)
        __builtin_nontemporal_store(acc[mb][nb][r] + bias,
                                    &xp[(size_t)(m0 + r) * k4H + n]);
    }
  }
}

// -------------------------------------------------------------------------
// Kernel 2: persistent recurrence. 64 WGs x 256 threads. WG g owns hidden
// units j in [g*16, g*16+16) -> gate columns {j, H+j, 2H+j, 3H+j} (64 cols).
// W_hh slice (64 x 1024, bf16, B-fragment-swizzled) lives in LDS (128 KB);
// cell state c (64x16 f32) lives in LDS. Per step: 64x64 GEMM tile over
// K=1024 via bf16 WMMA, gate nonlinearity, h write, grid barrier.
// -------------------------------------------------------------------------
__global__ void lstm_recurrence(const float* __restrict__ W_hh,
                                const float* __restrict__ xp,
                                __bf16* __restrict__ h0,
                                __bf16* __restrict__ h1,
                                unsigned int* __restrict__ cnt) {
  extern __shared__ char smem[];
  __bf16* whh    = (__bf16*)smem;                      // 128 KB frag store
  float*  gate_s = (float*)(smem + 131072);            // 16 KB [n_local*64+b]
  float*  c_s    = (float*)(smem + 131072 + 16384);    // 4 KB  [jl*64+b]

  const int g     = blockIdx.x;
  const int tid   = threadIdx.x;
  const int lane  = tid & 31;
  const int wave  = tid >> 5;
  const int lr    = lane & 15;
  const int half8 = (lane >> 4) * 8;

  // Stage W_hh slice -> LDS, pre-swizzled into B-fragment layout.
  // Fragment f = nb*32 + kb (nb in [0,4) gate, kb in [0,32) K-block).
  for (int f = wave * 16; f < wave * 16 + 16; ++f) {
    int nb = f >> 5;
    int kb = f & 31;
    int n  = nb * kH + g * 16 + lr;                    // global gate column
    const float* src = W_hh + (size_t)n * kH + kb * 32 + half8;
    __bf16* dst = whh + ((size_t)f * 32 + lane) * 16;
#pragma unroll
    for (int i = 0; i < 8; ++i) {
      dst[i]     = (__bf16)src[i];
      dst[i + 8] = (__bf16)src[i + 16];
    }
  }
  for (int i = tid; i < kB * 16; i += 256) c_s[i] = 0.0f;
  __syncthreads();

  // 16 C-fragments (4 M-blocks x 4 N-blocks) over 8 waves: 2 each.
  const int mb  = wave & 3;            // batch block (16 rows)
  const int nb0 = (wave >> 2) * 2;     // first of two gate blocks

  // Per-thread prefetch/elementwise chunk: batch pb, gate pg (64B chunk).
  const int pb = tid & 63;
  const int pg = tid >> 6;

  for (int t = 0; t < kT; ++t) {
    const __bf16* hprev = (t & 1) ? h1 : h0;
    __bf16*       hnext = (t & 1) ? h0 : h1;
    const float*  xpt   = xp + (size_t)t * kB * k4H;

    // Prefetch this step's xp slice (16 KB/WG, 64B chunk per thread) so the
    // WMMA K-loop hides its HBM latency.  -> global_prefetch_b8
    __builtin_prefetch(xpt + (size_t)pb * k4H + pg * kH + g * 16, 0, 0);

    v8f acc0 = vzero8(), acc1 = vzero8();
    const __bf16* arow = hprev + (size_t)(mb * 16 + lr) * kH + half8;
#pragma unroll 4
    for (int kb = 0; kb < 32; ++kb) {
      bf16x16 a   = load16_bf16(arow + kb * 32);
      bf16x16 bb0 = *(const bf16x16*)(whh + ((size_t)((nb0    ) * 32 + kb) * 32 + lane) * 16);
      bf16x16 bb1 = *(const bf16x16*)(whh + ((size_t)((nb0 + 1) * 32 + kb) * 32 + lane) * 16);
      acc0 = __builtin_amdgcn_wmma_f32_16x16x32_bf16(false, a, false, bb0,
                                                     (short)0, acc0, false, false);
      acc1 = __builtin_amdgcn_wmma_f32_16x16x32_bf16(false, a, false, bb1,
                                                     (short)0, acc1, false, false);
    }
    {
      int m0  = mb * 16 + half8;
      int nl0 = (nb0    ) * 16 + lr;
      int nl1 = (nb0 + 1) * 16 + lr;
#pragma unroll
      for (int r = 0; r < 8; ++r) {
        gate_s[nl0 * 64 + m0 + r] = acc0[r];
        gate_s[nl1 * 64 + m0 + r] = acc1[r];
      }
    }
    __syncthreads();

    // Elementwise LSTM cell for this WG's 16 hidden x 64 batch elements.
    // xp is streamed once -> non-temporal loads.
#pragma unroll
    for (int u = 0; u < 4; ++u) {
      int idx = tid * 4 + u;          // 1024 elements / 256 threads
      int b   = idx & 63;
      int jl  = idx >> 6;
      int j   = g * 16 + jl;
      const float* xpb = xpt + (size_t)b * k4H;
      float iv = gate_s[(     jl) * 64 + b] + __builtin_nontemporal_load(&xpb[j]);
      float fv = gate_s[(16 + jl) * 64 + b] + __builtin_nontemporal_load(&xpb[kH + j]);
      float gv = gate_s[(32 + jl) * 64 + b] + __builtin_nontemporal_load(&xpb[2 * kH + j]);
      float ov = gate_s[(48 + jl) * 64 + b] + __builtin_nontemporal_load(&xpb[3 * kH + j]);
      iv = sigmoidf(iv);
      fv = sigmoidf(fv);
      gv = tanhf(gv);
      ov = sigmoidf(ov);
      float c = fv * c_s[jl * 64 + b] + iv * gv;
      c_s[jl * 64 + b] = c;
      hnext[(size_t)b * kH + j] = (__bf16)(ov * tanhf(c));
    }

    // Grid-wide step barrier (per-step counter; release/acquire at DEV scope).
    __threadfence();
    __syncthreads();
    if (tid == 0) {
      __hip_atomic_fetch_add(&cnt[t], 1u, __ATOMIC_RELEASE,
                             __HIP_MEMORY_SCOPE_AGENT);
      while (__hip_atomic_load(&cnt[t], __ATOMIC_ACQUIRE,
                               __HIP_MEMORY_SCOPE_AGENT) < (unsigned)kNWG_REC)
        __builtin_amdgcn_s_sleep(1);
    }
    __syncthreads();
    __threadfence();
  }
}

// -------------------------------------------------------------------------
// Kernel 3: out[b] = sigmoid(h_final[b,:] . W_fc + b_fc)
// -------------------------------------------------------------------------
__global__ void lstm_fc_out(const __bf16* __restrict__ hfin,
                            const float* __restrict__ W_fc,
                            const float* __restrict__ b_fc,
                            float* __restrict__ out) {
  int b = threadIdx.x;
  if (b < kB) {
    float s = b_fc[0];
    for (int j = 0; j < kH; ++j)
      s += (float)hfin[(size_t)b * kH + j] * W_fc[j];
    out[b] = sigmoidf(s);
  }
}

extern "C" void kernel_launch(void* const* d_in, const int* in_sizes, int n_in,
                              void* d_out, int out_size, void* d_ws, size_t ws_size,
                              hipStream_t stream) {
  const int*   x    = (const int*)d_in[0];
  const float* emb  = (const float*)d_in[1];
  const float* W_ih = (const float*)d_in[2];
  const float* W_hh = (const float*)d_in[3];
  const float* b_ih = (const float*)d_in[4];
  const float* b_hh = (const float*)d_in[5];
  const float* W_fc = (const float*)d_in[6];
  const float* b_fc = (const float*)d_in[7];
  float* out = (float*)d_out;

  char* ws = (char*)d_ws;
  const size_t XP_BYTES = (size_t)kT * kB * k4H * sizeof(float);  // 512 MB
  const size_t H_BYTES  = (size_t)kB * kH * 2;                    // 128 KB bf16
  float*        xp  = (float*)ws;
  __bf16*       h0  = (__bf16*)(ws + XP_BYTES);
  __bf16*       h1  = (__bf16*)(ws + XP_BYTES + H_BYTES);
  unsigned int* cnt = (unsigned int*)(ws + XP_BYTES + 2 * H_BYTES);

  // h(t=0) = 0; per-step barrier counters = 0 (re-done on every launch/replay).
  hipMemsetAsync(h0, 0, H_BYTES, stream);
  hipMemsetAsync(cnt, 0, (size_t)kT * sizeof(unsigned int), stream);

  // Phase 1: all-timestep input projection GEMM (bf16 WMMA, f32 out).
  lstm_xproj_gemm<<<dim3(kB * kT / 64, k4H / 256), 256, 0, stream>>>(
      x, emb, W_ih, b_ih, b_hh, xp);

  // Phase 2: persistent recurrence with LDS-resident W_hh slices.
  const size_t REC_LDS = 131072 + 16384 + 4096;  // 148 KB of 320 KB/WGP
  hipFuncSetAttribute(reinterpret_cast<const void*>(lstm_recurrence),
                      hipFuncAttributeMaxDynamicSharedMemorySize, (int)REC_LDS);
  lstm_recurrence<<<kNWG_REC, 256, REC_LDS, stream>>>(W_hh, xp, h0, h1, cnt);

  // Phase 3: final FC + sigmoid. After 512 steps, final h lives in h0.
  lstm_fc_out<<<1, 64, 0, stream>>>(h0, W_fc, b_fc, out);
}
